// InfoNCELoss_39118562132130
// MI455X (gfx1250) — matrix operand order repaired and checked by reading, not compiled, not run on hardware
//
#include <hip/hip_runtime.h>
#include <hip/hip_bf16.h>

typedef __attribute__((ext_vector_type(16))) _Float16 v16h;
typedef __attribute__((ext_vector_type(8)))  _Float16 v8h;
typedef __attribute__((ext_vector_type(8)))  float    v8f;

#define NROWS 8192
#define DDIM  256
#define TILE_N 32
#define LDS_STRIDE 40   // halves per K-row: multiple of 8 keeps b128 16B-aligned,
                        // 80B row pitch spreads K-lanes across banks (<=2-way)

__device__ __forceinline__ float kInvT()  { return 1.0f / 0.07f; }
__device__ __forceinline__ float kLog2e() { return 1.4426950408889634f; }
__device__ __forceinline__ float kLn2()   { return 0.6931471805599453f; }

// ---------------------------------------------------------------------------
// Kernel 1: per-row L2 normalize, emit f16 anchor (row-major), f16 positive
// TRANSPOSED (Pt: [DDIM][NROWS]) so B-tile K-rows are contiguous along N, plus the
// exact fp32 diagonal cosine similarity diag[i].
// ---------------------------------------------------------------------------
__global__ void __launch_bounds__(256)
infonce_normalize(const float* __restrict__ anchor,
                  const float* __restrict__ positive,
                  _Float16* __restrict__ Ah,
                  _Float16* __restrict__ Pt,
                  float* __restrict__ diag) {
    __shared__ float red0[256];
    __shared__ float red1[256];
    __shared__ float red2[256];
    const int row = blockIdx.x;
    const int t   = threadIdx.x;

    const float av = anchor  [row * DDIM + t];
    const float pv = positive[row * DDIM + t];
    red0[t] = av * av;
    red1[t] = pv * pv;
    red2[t] = av * pv;
    __syncthreads();
    #pragma unroll
    for (int off = 128; off > 0; off >>= 1) {
        if (t < off) {
            red0[t] += red0[t + off];
            red1[t] += red1[t + off];
            red2[t] += red2[t + off];
        }
        __syncthreads();
    }
    const float na = fmaxf(sqrtf(red0[0]), 1e-12f);
    const float np = fmaxf(sqrtf(red1[0]), 1e-12f);

    Ah[row * DDIM + t]  = (_Float16)(av / na);
    Pt[t * NROWS + row] = (_Float16)(pv / np);
    if (t == 0) diag[row] = red2[0] / (na * np);
}

// ---------------------------------------------------------------------------
// Kernel 2: streaming sim = Ah x Pt^T with LDS-staged, double-buffered B tiles.
// Block = 4 waves = 64 rows; per iteration the block stages a 256Kx32N f16
// tile of Pt into LDS (shared by all 4 waves -> 4x global traffic reduction).
// Global loads for tile t+1 are issued BEFORE consuming tile t, and the
// LDS stores happen after, so global latency hides behind 16 WMMAs.
//
// A fragment (16-bit A 16x32, ISA 7.12.2): lane L holds row M=L&15, two
// contiguous b128 half-groups. B fragment (32x16): lane holds N=L&15, 16 K halves in A's k order
// -> 16 strided 16-bit reads from the padded LDS tile. C layout: lanes 0-15
// hold M=vg, lanes 16-31 hold M=8+vg, N = lane&15.
// ---------------------------------------------------------------------------
__global__ void __launch_bounds__(128)
infonce_simloss(const _Float16* __restrict__ Ah,
                const _Float16* __restrict__ Pt,
                const float* __restrict__ diag,
                float* __restrict__ loss) {
    __shared__ _Float16 tile[2][DDIM * LDS_STRIDE];  // 2 x 20KB

    const int tid  = threadIdx.x;
    const int lane = tid & 31;
    const int wave = tid >> 5;
    const int r0   = blockIdx.x * 64 + wave * 16;
    const int mlow = lane & 15;
    const int hi   = lane >> 4;

    // ---- Resident A fragments for all 8 K-steps (K = 256 = 8 * 32) ----
    v16h afrag[8];
    const _Float16* arow = Ah + (r0 + mlow) * DDIM;
    #pragma unroll
    for (int ks = 0; ks < 8; ++ks) {
        const int kbase = ks * 32;
        const v8h lo = *(const v8h*)(arow + kbase + (hi ? 8 : 0));
        const v8h hh = *(const v8h*)(arow + kbase + (hi ? 24 : 16));
        #pragma unroll
        for (int e = 0; e < 8; ++e) { afrag[ks][e] = lo[e]; afrag[ks][8 + e] = hh[e]; }
    }

    const float KL = kLog2e() * kInvT();  // logits in log2 domain
    const float M2 = KL;                  // fixed bound: cosine sim <= 1
    float s[8], offm[8];
    #pragma unroll
    for (int vg = 0; vg < 8; ++vg) { s[vg] = 0.0f; offm[vg] = -3.0e38f; }

    // Staging: 256 K-rows x 32 N halves = 1024 b128 chunks, 8 per thread.
    // chunk c: k = c>>2, q = c&3; global 16B at Pt[k][n0 + 8q],
    // LDS 16B at tile[buf][k*LDS_STRIDE + 8q].
    v8h stg[8];

    // Preload + store tile 0.
    #pragma unroll
    for (int j = 0; j < 8; ++j) {
        const int c = tid + 128 * j, k = c >> 2, q = c & 3;
        stg[j] = *(const v8h*)(Pt + (size_t)k * NROWS + 8 * q);
    }
    #pragma unroll
    for (int j = 0; j < 8; ++j) {
        const int c = tid + 128 * j, k = c >> 2, q = c & 3;
        *(v8h*)(&tile[0][k * LDS_STRIDE + 8 * q]) = stg[j];
    }
    __syncthreads();

    const int NTILES = NROWS / TILE_N;
    for (int t = 0; t < NTILES; ++t) {
        const int buf = t & 1;
        const int n0  = t * TILE_N;
        const bool more = (t + 1) < NTILES;

        // Issue next tile's global loads now; consumed ~16 WMMAs later.
        if (more) {
            #pragma unroll
            for (int j = 0; j < 8; ++j) {
                const int c = tid + 128 * j, k = c >> 2, q = c & 3;
                stg[j] = *(const v8h*)(Pt + (size_t)k * NROWS + (n0 + TILE_N) + 8 * q);
            }
        }

        // Consume two 16-column subtiles from LDS.
        #pragma unroll
        for (int ns = 0; ns < 2; ++ns) {
            const _Float16* lbase = &tile[buf][(hi * 8) * LDS_STRIDE + ns * 16 + mlow];
            v16h bf[2];
            {   // fragment ks = 0
                #pragma unroll
                for (int e = 0; e < 8; ++e) { bf[0][e] = lbase[e * LDS_STRIDE]; bf[0][8 + e] = lbase[(16 + e) * LDS_STRIDE]; }
            }
            v8f c = {};
            #pragma unroll
            for (int ks = 0; ks < 8; ++ks) {
                if (ks < 7) {  // pipeline: fetch fragment ks+1 during WMMA ks
                    const _Float16* lk = lbase + (ks + 1) * 32 * LDS_STRIDE;
                    #pragma unroll
                    for (int e = 0; e < 8; ++e) {
                        bf[(ks + 1) & 1][e] = lk[e * LDS_STRIDE];
                        bf[(ks + 1) & 1][8 + e] = lk[(16 + e) * LDS_STRIDE];
                    }
                }
                c = __builtin_amdgcn_wmma_f32_16x16x32_f16(
                        false, afrag[ks], false, bf[ks & 1],
                        (short)0, c, false, false);
            }

            // Fused epilogue: one exp2 + one masked max per sim element.
            #pragma unroll
            for (int vg = 0; vg < 8; ++vg) {
                const int rowg = r0 + hi * 8 + vg;
                const int col  = n0 + ns * 16 + mlow;
                const float v2 = c[vg] * KL;
                s[vg] += exp2f(v2 - M2);
                if (col != rowg) offm[vg] = fmaxf(offm[vg], v2);
            }
        }

        // Store next tile to the other LDS buffer (waits on global loads here,
        // after the WMMA work, not before it).
        if (more) {
            #pragma unroll
            for (int j = 0; j < 8; ++j) {
                const int c = tid + 128 * j, k = c >> 2, q = c & 3;
                *(v8h*)(&tile[buf ^ 1][k * LDS_STRIDE + 8 * q]) = stg[j];
            }
        }
        __syncthreads();
    }

    // Reduce across the 16 lanes of each half (xor masks stay within a half).
    #pragma unroll
    for (int m = 8; m > 0; m >>= 1) {
        #pragma unroll
        for (int vg = 0; vg < 8; ++vg) {
            s[vg]   += __shfl_xor(s[vg], m, 32);
            offm[vg] = fmaxf(offm[vg], __shfl_xor(offm[vg], m, 32));
        }
    }

    if (mlow == 0) {
        #pragma unroll
        for (int vg = 0; vg < 8; ++vg) {
            const int rowg  = r0 + hi * 8 + vg;
            const float tot = s[vg] + exp2f(offm[vg] - M2);  // + hard-negative logit
            const float lse = kLn2() * (M2 + log2f(tot));    // natural-log LSE
            loss[rowg] = lse - diag[rowg] * kInvT();
        }
    }
}

// ---------------------------------------------------------------------------
// Kernel 3: deterministic mean reduction (no float atomics).
// ---------------------------------------------------------------------------
__global__ void __launch_bounds__(256)
infonce_reduce(const float* __restrict__ loss, float* __restrict__ out) {
    __shared__ float red[256];
    const int t = threadIdx.x;
    float acc = 0.0f;
    for (int i = t; i < NROWS; i += 256) acc += loss[i];
    red[t] = acc;
    __syncthreads();
    #pragma unroll
    for (int off = 128; off > 0; off >>= 1) {
        if (t < off) red[t] += red[t + off];
        __syncthreads();
    }
    if (t == 0) out[0] = red[0] / (float)NROWS;
}

extern "C" void kernel_launch(void* const* d_in, const int* in_sizes, int n_in,
                              void* d_out, int out_size, void* d_ws, size_t ws_size,
                              hipStream_t stream) {
    const float* anchor   = (const float*)d_in[0];
    const float* positive = (const float*)d_in[1];

    char* ws = (char*)d_ws;
    _Float16* Ah   = (_Float16*)(ws);                                   // 4 MiB
    _Float16* Pt   = (_Float16*)(ws + (size_t)4 * 1024 * 1024);         // 4 MiB
    float*    diag = (float*)   (ws + (size_t)8 * 1024 * 1024);         // 32 KiB
    float*    loss = (float*)   (ws + (size_t)8 * 1024 * 1024 + 32768); // 32 KiB

    infonce_normalize<<<NROWS, 256, 0, stream>>>(anchor, positive, Ah, Pt, diag);
    infonce_simloss<<<NROWS / 64, 128, 0, stream>>>(Ah, Pt, diag, loss);
    infonce_reduce<<<1, 256, 0, stream>>>(loss, (float*)d_out);
}
